// QTransformerTorchGen136_65481071395354
// MI455X (gfx1250) — matrix operand order repaired and checked by reading, not compiled, not run on hardware
//
#include <hip/hip_runtime.h>
#include <hip/hip_bf16.h>
#include <math.h>

// Problem dims (from reference)
#define NB   16
#define NS   2048
#define NE   256
#define NL   2
#define NNQ  8
#define NF   512
#define NC   2
#define NDK  64

#define MROWS 32   // rows per block (M-block), 2 WMMA M-tiles

typedef __attribute__((ext_vector_type(16))) _Float16 v16h;
typedef __attribute__((ext_vector_type(8)))  float    v8f;
typedef __attribute__((ext_vector_type(4)))  float    f32x4;

union HalfVec { v16h v; _Float16 h[16]; f32x4 f4[2]; };

// A-fragment (16x32 f16, row-major tile with leading dim ld), per CDNA5 ISA layout:
// lanes 0-15 hold row=lane, K {0..7,16..23}; lanes 16-31 hold row=lane-16, K {8..15,24..31}
__device__ inline v16h load_a_frag(const _Float16* tile, int ld, int lane) {
  const int r = lane & 15;
  const int c = (lane >> 4) << 3;          // 0 or 8
  const _Float16* p = tile + r * ld + c;
  HalfVec u;
  u.f4[0] = *(const f32x4*)(p);            // K = c .. c+7
  u.f4[1] = *(const f32x4*)(p + 16);       // K = c+16 .. c+23
  return u.v;
}

// B-fragment (32x16 f16) from a pre-transposed weight wT[N][K] (row-major, ldk):
// lane n (0-15) holds column n0+n, K = k0+{0..7,16..23}; lanes 16-31 hold K = k0+{8..15,24..31}
__device__ inline v16h load_b_frag(const _Float16* wT, int ldk, int n0, int k0, int lane) {
  const int n = n0 + (lane & 15);
  const int k = k0 + ((lane >> 4) << 3);   // 0 or +8
  const _Float16* p = wT + n * ldk + k;
  HalfVec u;
  u.f4[0] = *(const f32x4*)(p);
  u.f4[1] = *(const f32x4*)(p + 16);
  return u.v;
}

__device__ inline float wave_sum(float v) {
#pragma unroll
  for (int off = 16; off > 0; off >>= 1) v += __shfl_xor(v, off, 32);
  return v;
}

// 2Mx2N register-tiled GEMM: D[32 x 256] = A[32 x K] * B^T-layout[256][K]
// wave w owns N-tiles 2w, 2w+1 (disjoint across waves -> B read once per block)
// and both M-tiles (B fragments reused across M).
__device__ inline void gemm_2x2(const _Float16* __restrict__ aBase, int lda, int K,
                                const _Float16* __restrict__ wT,
                                float (*outs)[NE], int wid, int lane) {
  v8f acc00 = {}, acc01 = {}, acc10 = {}, acc11 = {};
  const int n0 = (wid * 2) * 16;
  const int n1 = n0 + 16;
#pragma unroll
  for (int kk = 0; kk < 512; kk += 32) {      // bounded by runtime K below
    if (kk >= K) break;
    const v16h a0 = load_a_frag(aBase + kk, lda, lane);            // M-tile 0
    const v16h a1 = load_a_frag(aBase + 16 * lda + kk, lda, lane); // M-tile 1
    const v16h b0 = load_b_frag(wT, K, n0, kk, lane);
    acc00 = __builtin_amdgcn_wmma_f32_16x16x32_f16(false, a0, false, b0,
                                                   (short)0, acc00, false, false);
    acc10 = __builtin_amdgcn_wmma_f32_16x16x32_f16(false, a1, false, b0,
                                                   (short)0, acc10, false, false);
    const v16h b1 = load_b_frag(wT, K, n1, kk, lane);
    acc01 = __builtin_amdgcn_wmma_f32_16x16x32_f16(false, a0, false, b1,
                                                   (short)0, acc01, false, false);
    acc11 = __builtin_amdgcn_wmma_f32_16x16x32_f16(false, a1, false, b1,
                                                   (short)0, acc11, false, false);
  }
  // C/D layout: element j of acc -> row = j + 8*(lane>>4), col = lane&15
  const int row0 = (lane >> 4) << 3;
  const int col  = lane & 15;
#pragma unroll
  for (int j = 0; j < 8; ++j) {
    outs[row0 + j][n0 + col]      = acc00[j];
    outs[row0 + j][n1 + col]      = acc01[j];
    outs[16 + row0 + j][n0 + col] = acc10[j];
    outs[16 + row0 + j][n1 + col] = acc11[j];
  }
}

// residual + bias + LayerNorm over E=256, 4 rows per wave, result written back to xs
__device__ inline void resid_ln(float (*xs)[NE], float (*outs)[NE],
                                const float* __restrict__ bias,
                                const float* __restrict__ g,
                                const float* __restrict__ b,
                                int wid, int lane) {
#pragma unroll
  for (int rr = 0; rr < MROWS / 8; ++rr) {
    const int r = wid * (MROWS / 8) + rr;
    float v[8];
    float s = 0.f;
#pragma unroll
    for (int ii = 0; ii < 8; ++ii) {
      const int e = lane + 32 * ii;
      v[ii] = xs[r][e] + outs[r][e] + bias[e];
      s += v[ii];
    }
    const float mean = wave_sum(s) * (1.0f / NE);
    float vs = 0.f;
#pragma unroll
    for (int ii = 0; ii < 8; ++ii) { const float d = v[ii] - mean; vs += d * d; }
    const float var  = wave_sum(vs) * (1.0f / NE);
    const float rstd = rsqrtf(var + 1e-5f);
#pragma unroll
    for (int ii = 0; ii < 8; ++ii) {
      const int e = lane + 32 * ii;
      xs[r][e] = (v[ii] - mean) * rstd * g[e] + b[e];
    }
  }
}

// ---------------------------------------------------------------------------
// Weight prep: f32 -> f16 with transpose so WMMA B-fragments are contiguous.
//   wcT[l][n][k] = combine_w[l][k][n]   (256x256 per layer)
//   w2T[l][e][f] = lin2_w[l][f][e]      (256x512 per layer)
// Also zeros the pooled accumulator (re-run every launch -> deterministic).
// ---------------------------------------------------------------------------
__global__ void prep_weights(const float* __restrict__ combine_w,
                             const float* __restrict__ lin2_w,
                             _Float16* __restrict__ wcT,
                             _Float16* __restrict__ w2T,
                             float* __restrict__ pooled) {
  const int stride = gridDim.x * blockDim.x;
  const int t0 = blockIdx.x * blockDim.x + threadIdx.x;
  for (int idx = t0; idx < NL * NE * NE; idx += stride) {
    const int l = idx / (NE * NE);
    const int r = idx - l * NE * NE;
    const int n = r / NE, k = r - n * NE;
    wcT[idx] = (_Float16)combine_w[l * NE * NE + k * NE + n];
  }
  for (int idx = t0; idx < NL * NE * NF; idx += stride) {
    const int l = idx / (NE * NF);
    const int r = idx - l * NE * NF;
    const int e = r / NF, f = r - e * NF;
    w2T[idx] = (_Float16)lin2_w[l * NF * NE + f * NE + e];
  }
  for (int idx = t0; idx < NB * NE; idx += stride) pooled[idx] = 0.f;
}

// ---------------------------------------------------------------------------
// Fused forward: each block owns MROWS=32 token rows, kept resident in LDS
// through both layers; WMMA f16->f32 for both GEMMs; pooled via atomicAdd.
// ---------------------------------------------------------------------------
__global__ __launch_bounds__(256, 1) void qtx_forward(
    const int*   __restrict__ tokens,
    const float* __restrict__ embed,
    const float* __restrict__ attn_theta,
    const float* __restrict__ combine_b,
    const float* __restrict__ ffn_theta,
    const float* __restrict__ lin1_w,
    const float* __restrict__ lin1_b,
    const float* __restrict__ lin2_b,
    const float* __restrict__ ln1_g, const float* __restrict__ ln1_b,
    const float* __restrict__ ln2_g, const float* __restrict__ ln2_b,
    const _Float16* __restrict__ wcT,
    const _Float16* __restrict__ w2T,
    float* __restrict__ pooled) {
  __shared__ __align__(16) float    xs[MROWS][NE];     // 32 KB activation block
  __shared__ __align__(16) float    outs[MROWS][NE];   // 32 KB GEMM output
  __shared__ __align__(16) _Float16 hbuf[MROWS * NF];  // 32 KB: A-tile / FFN hidden
  __shared__ __align__(16) float    w1s[NNQ * NF];     // 16 KB lin1 weights
  __shared__ float b1s[NF];                            //  2 KB lin1 bias
  __shared__ float qs[MROWS][NNQ];                     //  1 KB

  const int tid  = threadIdx.x;
  const int lane = tid & 31;
  const int wid  = tid >> 5;
  const int m0   = blockIdx.x * MROWS;

  // ---- token embedding + sinusoidal positional encoding ----
  const float kfreq = -9.210340371976184f / (float)NE;  // -ln(10000)/E
  {
    const int e = tid;
    const float freq = __expf(kfreq * (float)(e & ~1));
#pragma unroll
    for (int r = 0; r < MROWS; ++r) {
      const int m = m0 + r;
      const int s = m & (NS - 1);
      const float arg = (float)s * freq;
      const float pe  = (e & 1) ? __cosf(arg) : __sinf(arg);
      xs[r][e] = embed[tokens[m] * NE + e] + pe;
    }
  }
  __syncthreads();

  for (int l = 0; l < NL; ++l) {
    // ---- quantum attention encode: A = cos(x + tile(theta,H)) as f16 ----
    _Float16* atile = hbuf;  // [MROWS][NE]
    {
      const int e = tid;
      const float th = attn_theta[l * NDK + (e & (NDK - 1))];
#pragma unroll
      for (int r = 0; r < MROWS; ++r)
        atile[r * NE + e] = (_Float16)__cosf(xs[r][e] + th);
      // stage lin1 weights/bias for the VALU FFN stage
#pragma unroll
      for (int j = 0; j < (NNQ * NF) / 256; ++j)
        w1s[tid + 256 * j] = lin1_w[l * NNQ * NF + tid + 256 * j];
      b1s[tid]       = lin1_b[l * NF + tid];
      b1s[tid + 256] = lin1_b[l * NF + tid + 256];
    }
    __syncthreads();

    // ---- GEMM1: [32x256] x [256x256], 2Mx2N per wave ----
    gemm_2x2(atile, NE, NE, wcT + l * NE * NE, outs, wid, lane);
    __syncthreads();

    // ---- x = LN1(x + attn + bias) ----
    resid_ln(xs, outs, combine_b + l * NE, ln1_g + l * NE, ln1_b + l * NE, wid, lane);
    __syncthreads();

    // ---- FFN quantum encode: q = cos(x[:, :8]) * cos(theta) ----
    {
      const int r = tid >> 3, k = tid & 7;      // 32*8 == 256 threads
      qs[r][k] = __cosf(xs[r][k]) * __cosf(ffn_theta[l * NNQ + k]);
    }
    __syncthreads();

    // ---- h = relu(q @ W1 + b1) (K=8: VALU from LDS, result f16 in LDS) ----
#pragma unroll
    for (int j = 0; j < (MROWS * NF) / 256; ++j) {
      const int i = tid + 256 * j;
      const int r = i >> 9;
      const int f = i & (NF - 1);
      float acc = b1s[f];
#pragma unroll
      for (int k = 0; k < NNQ; ++k) acc += qs[r][k] * w1s[k * NF + f];
      hbuf[r * NF + f] = (_Float16)fmaxf(acc, 0.f);
    }
    __syncthreads();

    // ---- GEMM2: [32x512] x [512x256], 2Mx2N per wave ----
    gemm_2x2(hbuf, NF, NF, w2T + l * NE * NF, outs, wid, lane);
    __syncthreads();

    // ---- x = LN2(x + ffn + bias) ----
    resid_ln(xs, outs, lin2_b + l * NE, ln2_g + l * NE, ln2_b + l * NE, wid, lane);
    __syncthreads();
  }

  // ---- partial mean-pool: all MROWS rows of this block belong to one batch b ----
  {
    const int b = m0 >> 11;  // m0 / S
    float s = 0.f;
#pragma unroll
    for (int r = 0; r < MROWS; ++r) s += xs[r][tid];
    atomicAdd(&pooled[b * NE + tid], s);
  }
}

// ---------------------------------------------------------------------------
// logits[b][c] = (pooled[b]/S) @ cls_w + cls_b  (16x2 outputs, trivial)
// ---------------------------------------------------------------------------
__global__ void classify(const float* __restrict__ pooled,
                         const float* __restrict__ cls_w,
                         const float* __restrict__ cls_b,
                         float* __restrict__ out) {
  const int t = threadIdx.x;
  if (t < NB * NC) {
    const int b = t / NC, c = t - b * NC;
    float acc = cls_b[c];
    for (int e = 0; e < NE; ++e)
      acc += (pooled[b * NE + e] * (1.0f / NS)) * cls_w[e * NC + c];
    out[b * NC + c] = acc;
  }
}

extern "C" void kernel_launch(void* const* d_in, const int* in_sizes, int n_in,
                              void* d_out, int out_size, void* d_ws, size_t ws_size,
                              hipStream_t stream) {
  (void)in_sizes; (void)n_in; (void)out_size; (void)ws_size;
  const int*   tokens     = (const int*)  d_in[0];
  const float* embed      = (const float*)d_in[1];
  const float* attn_theta = (const float*)d_in[2];
  const float* combine_w  = (const float*)d_in[3];
  const float* combine_b  = (const float*)d_in[4];
  const float* ffn_theta  = (const float*)d_in[5];
  const float* lin1_w     = (const float*)d_in[6];
  const float* lin1_b     = (const float*)d_in[7];
  const float* lin2_w     = (const float*)d_in[8];
  const float* lin2_b     = (const float*)d_in[9];
  const float* ln1_g      = (const float*)d_in[10];
  const float* ln1_b      = (const float*)d_in[11];
  const float* ln2_g      = (const float*)d_in[12];
  const float* ln2_b      = (const float*)d_in[13];
  const float* cls_w      = (const float*)d_in[14];
  const float* cls_b      = (const float*)d_in[15];
  float* out = (float*)d_out;

  char* ws = (char*)d_ws;
  _Float16* wcT = (_Float16*)ws;                                   // 256 KB
  _Float16* w2T = (_Float16*)(ws + (size_t)NL * NE * NE * 2);      // 512 KB
  float* pooled = (float*)(ws + (size_t)NL * NE * NE * 2
                              + (size_t)NL * NE * NF * 2);         // 16 KB

  prep_weights<<<512, 256, 0, stream>>>(combine_w, lin2_w, wcT, w2T, pooled);
  qtx_forward<<<(NB * NS) / MROWS, 256, 0, stream>>>(
      tokens, embed, attn_theta, combine_b, ffn_theta, lin1_w, lin1_b,
      lin2_b, ln1_g, ln1_b, ln2_g, ln2_b, wcT, w2T, pooled);
  classify<<<1, 32, 0, stream>>>(pooled, cls_w, cls_b, out);
}
